// OverlapNAT_83854941487366
// MI455X (gfx1250) — matrix operand order repaired and checked by reading, not compiled, not run on hardware
//
#include <hip/hip_runtime.h>

typedef __attribute__((ext_vector_type(16))) _Float16 v16h;
typedef __attribute__((ext_vector_type(8)))  _Float16 v8h;
typedef __attribute__((ext_vector_type(8)))  float    v8f;

// ---------------- problem constants ----------------
#define BATCH  2
#define CIN    3
#define HIMG   256
#define DMODEL 128
#define HP     129                 // feature map H = W
#define NT     (BATCH*HP*HP)       // 33282 real tokens
#define NTPAD  33312               // 32*1041
#define HA     133                 // wrap-padded attention map (HP + 2*2)
#define NA     (BATCH*HA*HA)       // 35378
#define NAPAD  35392               // 32*1106
#define NHEADS 8
#define DH     16

// =====================================================================
// f32 -> f16 weight conversion (once per launch, tiny)
// =====================================================================
__global__ __launch_bounds__(256) void cvt_f16_kernel(
    const float* __restrict__ src, _Float16* __restrict__ dst, int n)
{
    int i = blockIdx.x * blockDim.x + threadIdx.x;
    if (i < n) dst[i] = (_Float16)src[i];
}

// =====================================================================
// WMMA GEMM:  Y[M x Nout] (+)= X[M x K] * W[Nout x K]^T + bias
// f16 operands, f32 accumulate.  One wave computes a 32x64 output tile:
// 2 A-fragments x 4 B-fragments = 8 v_wmma per 32-wide K step.
// blockDim = 64 (2 waves -> 32x128 per block).  grid = (Mpad/32, Nout/128).
// Fragments are loaded from global directly in contiguous k order (k = 16*hi + s
// per lane, same for A and B); all loads are 16B/32B aligned -> b128.
// addto=1: accumulate into f32 buffer Yf (residual); else write f16 Yh.
// =====================================================================
__global__ __launch_bounds__(64) void gemm_wmma_kernel(
    const _Float16* __restrict__ X, const _Float16* __restrict__ W,
    const float* __restrict__ bias,
    float* __restrict__ Yf, _Float16* __restrict__ Yh,
    int K, int Nout, int act, int addto)
{
    const int lane  = threadIdx.x & 31;
    const int wave  = threadIdx.x >> 5;
    const int m0    = blockIdx.x * 32;
    const int n0    = blockIdx.y * 128 + wave * 64;
    const int hi    = (lane >= 16);
    const int nlane = lane & 15;
    const int kbA   = hi ? 16 : 0;   // A-fragment K sub-offset (halves)
    const int kbB   = hi ? 16 : 0;   // B-fragment K sub-offset (halves)

    const _Float16* Xr0 = X + (size_t)(m0 + nlane) * K + kbA;       // A rows m0..m0+15
    const _Float16* Xr1 = Xr0 + (size_t)16 * K;                     // A rows m0+16..m0+31
    const _Float16* W0  = W + (size_t)(n0 + nlane) * K + kbB;       // B col N = lane&15

    v8f acc[2][4];
    #pragma unroll
    for (int s = 0; s < 2; ++s)
        #pragma unroll
        for (int t = 0; t < 4; ++t) acc[s][t] = v8f{};

    for (int k0 = 0; k0 < K; k0 += 32) {
        v16h a0, a1;
        {
            v8h lo = *(const v8h*)(Xr0 + k0);
            v8h hh = *(const v8h*)(Xr0 + k0 + 8);
            #pragma unroll
            for (int i = 0; i < 8; ++i) { a0[i] = lo[i]; a0[8 + i] = hh[i]; }
            lo = *(const v8h*)(Xr1 + k0);
            hh = *(const v8h*)(Xr1 + k0 + 8);
            #pragma unroll
            for (int i = 0; i < 8; ++i) { a1[i] = lo[i]; a1[8 + i] = hh[i]; }
        }
        #pragma unroll
        for (int t = 0; t < 4; ++t) {
            v16h b = *(const v16h*)(W0 + (size_t)t * 16 * K + k0);
            acc[0][t] = __builtin_amdgcn_wmma_f32_16x16x32_f16(
                false, a0, false, b, (short)0, acc[0][t], false, false);
            acc[1][t] = __builtin_amdgcn_wmma_f32_16x16x32_f16(
                false, a1, false, b, (short)0, acc[1][t], false, false);
        }
    }

    // ---- store: C/D layout: VGPR r -> M = r + (hi?8:0), N = lane&15
    const int moff = hi ? 8 : 0;
    #pragma unroll
    for (int t = 0; t < 4; ++t) {
        const int n = n0 + t * 16 + nlane;
        const float bv = bias[n];
        #pragma unroll
        for (int s = 0; s < 2; ++s) {
            #pragma unroll
            for (int r = 0; r < 8; ++r) {
                const int m = m0 + s * 16 + moff + r;
                const size_t idx = (size_t)m * Nout + n;
                float v = acc[s][t][r] + bv;
                if (act) v = v > 0.f ? v : 0.01f * v;
                if (addto) Yf[idx] += v;
                else       Yh[idx] = (_Float16)v;
            }
        }
    }
}

// =====================================================================
// LayerNorm over 128 channels (f32 in, f16 out); optionally writes the
// wrap-padded (133x133) token map so QKV becomes a dense GEMM.
// One block (128 thr) per destination row.
// =====================================================================
__global__ __launch_bounds__(128) void ln_kernel(
    const float* __restrict__ src, _Float16* __restrict__ dst,
    const float* __restrict__ g, const float* __restrict__ bta, int padded)
{
    __shared__ float red[128];
    const int t = blockIdx.x;
    const int d = threadIdx.x;
    int srow;
    if (padded) {
        if (t >= NA) srow = 0;
        else {
            int b = t / (HA * HA);
            int rem = t - b * HA * HA;
            int yp = rem / HA, xp = rem - yp * HA;
            int ys = yp - 2; if (ys < 0) ys += HP; if (ys >= HP) ys -= HP;
            int xs = xp - 2; if (xs < 0) xs += HP; if (xs >= HP) xs -= HP;
            srow = b * (HP * HP) + ys * HP + xs;
        }
    } else {
        srow = (t < NT) ? t : 0;
    }
    const float v = src[(size_t)srow * DMODEL + d];
    red[d] = v; __syncthreads();
    #pragma unroll
    for (int s = 64; s > 0; s >>= 1) { if (d < s) red[d] += red[d + s]; __syncthreads(); }
    const float mean = red[0] * (1.f / 128.f);
    __syncthreads();
    const float c = v - mean;
    red[d] = c * c; __syncthreads();
    #pragma unroll
    for (int s = 64; s > 0; s >>= 1) { if (d < s) red[d] += red[d + s]; __syncthreads(); }
    const float var = red[0] * (1.f / 128.f);
    dst[(size_t)t * DMODEL + d] = (_Float16)(c * rsqrtf(var + 1e-5f) * g[d] + bta[d]);
}

// =====================================================================
// 5x5 neighborhood attention, one thread per (token, head).
// qkv rows (f16) live in the padded (133x133) token space, layout [row][384].
// =====================================================================
__global__ __launch_bounds__(256) void attn_kernel(
    const _Float16* __restrict__ qkv, const float* __restrict__ rpb,
    _Float16* __restrict__ aout)
{
    const int idx = blockIdx.x * blockDim.x + threadIdx.x;
    if (idx >= NT * NHEADS) return;
    const int h   = idx & 7;
    const int tok = idx >> 3;
    const int b   = tok / (HP * HP);
    const int rem = tok - b * HP * HP;
    const int y   = rem / HP, x = rem - y * HP;
    const int baseA = b * (HA * HA);

    float q[DH];
    {
        const _Float16* qp = qkv + (size_t)(baseA + (y + 2) * HA + (x + 2)) * 384 + h * DH;
        #pragma unroll
        for (int d = 0; d < DH; ++d) q[d] = (float)qp[d] * 0.25f;   // dh^-0.5
    }
    float logit[25];
    float mx = -1e30f;
    #pragma unroll
    for (int i = 0; i < 5; ++i)
        #pragma unroll
        for (int j = 0; j < 5; ++j) {
            const _Float16* kp = qkv + (size_t)(baseA + (y + i) * HA + (x + j)) * 384 + 128 + h * DH;
            float s = rpb[(h * 9 + (2 + i)) * 9 + (2 + j)];
            #pragma unroll
            for (int d = 0; d < DH; ++d) s += q[d] * (float)kp[d];
            logit[i * 5 + j] = s;
            mx = fmaxf(mx, s);
        }
    float sum = 0.f;
    #pragma unroll
    for (int n = 0; n < 25; ++n) { float e = __expf(logit[n] - mx); logit[n] = e; sum += e; }
    const float inv = 1.f / sum;
    float o[DH];
    #pragma unroll
    for (int d = 0; d < DH; ++d) o[d] = 0.f;
    #pragma unroll
    for (int i = 0; i < 5; ++i)
        #pragma unroll
        for (int j = 0; j < 5; ++j) {
            const _Float16* vp = qkv + (size_t)(baseA + (y + i) * HA + (x + j)) * 384 + 256 + h * DH;
            const float a = logit[i * 5 + j] * inv;
            #pragma unroll
            for (int d = 0; d < DH; ++d) o[d] += a * (float)vp[d];
        }
    _Float16* op = aout + (size_t)tok * DMODEL + h * DH;
    #pragma unroll
    for (int d = 0; d < DH; ++d) op[d] = (_Float16)o[d];
}

// =====================================================================
// Patch embed: x*pos_w+pos_b (inline), wrap-pad 2, 4x4 conv stride 2 -> f (f32)
// One block (128 thr) per token; 48 im2col values staged in LDS.
// =====================================================================
__global__ __launch_bounds__(128) void patch_conv_kernel(
    const float* __restrict__ x, const float* __restrict__ pw,
    const float* __restrict__ pb, const float* __restrict__ wgt,
    const float* __restrict__ bias, float* __restrict__ f)
{
    const int tok = blockIdx.x;
    const int co  = threadIdx.x;
    if (tok >= NT) { f[(size_t)tok * DMODEL + co] = 0.f; return; }   // zero pad rows
    __shared__ float s[48];
    const int b = tok / (HP * HP);
    const int rem = tok - b * HP * HP;
    const int oy = rem / HP, ox = rem - oy * HP;
    if (co < 48) {
        const int c = co / 16, kk = co - c * 16;
        const int kh = kk / 4, kw = kk - kh * 4;
        const int hh = (2 * oy + kh - 2 + 256) & 255;
        const int ww = (2 * ox + kw - 2 + 256) & 255;
        const size_t pi = ((size_t)c * 256 + hh) * 256 + ww;
        const size_t xi = ((size_t)(b * 3 + c) * 256 + hh) * 256 + ww;
        s[co] = x[xi] * pw[pi] + pb[pi];
    }
    __syncthreads();
    float acc = bias[co];
    #pragma unroll
    for (int q = 0; q < 48; ++q) acc += s[q] * wgt[co * 48 + q];
    f[(size_t)tok * DMODEL + co] = acc;
}

// =====================================================================
// Depatchify: transposed conv (stride 2, 4x4) of f (129x129x128, f32)
// -> y(2,3,256,256)
// =====================================================================
__global__ __launch_bounds__(256) void depatch_kernel(
    const float* __restrict__ f, const float* __restrict__ w,
    const float* __restrict__ bias, float* __restrict__ y)
{
    const int idx = blockIdx.x * blockDim.x + threadIdx.x;
    if (idx >= BATCH * 3 * 256 * 256) return;
    const int wc = idx & 255;
    int t = idx >> 8;
    const int hh = t & 255; t >>= 8;
    const int c = t % 3, b = t / 3;
    const int oy = hh + 2, ox = wc + 2;       // cropped [2,258) of 260-wide output
    float acc = bias[c];
    for (int kh = 0; kh < 4; ++kh) {
        const int u = oy + kh - 3;
        if (u & 1) continue;
        const int iy = u >> 1;
        if (iy < 0 || iy > 128) continue;
        for (int kw = 0; kw < 4; ++kw) {
            const int v = ox + kw - 3;
            if (v & 1) continue;
            const int ix = v >> 1;
            if (ix < 0 || ix > 128) continue;
            const float* fp = f + (size_t)(b * (HP * HP) + iy * HP + ix) * DMODEL;
            const int woff = (c * 4 + (3 - kh)) * 4 + (3 - kw);   // depatch_w[d][c][3-kh][3-kw]
            float a = 0.f;
            #pragma unroll 8
            for (int d = 0; d < DMODEL; ++d) a += fp[d] * w[(size_t)d * 48 + woff];
            acc += a;
        }
    }
    y[idx] = acc;
}

// =====================================================================
// Residual head: 5x5 wrap conv 3->12 + leaky_relu, then 1x1 12->3 + add y
// =====================================================================
__global__ __launch_bounds__(256) void res1_kernel(
    const float* __restrict__ y, const float* __restrict__ w1,
    const float* __restrict__ b1, float* __restrict__ r1)
{
    const int idx = blockIdx.x * blockDim.x + threadIdx.x;
    if (idx >= BATCH * 12 * 256 * 256) return;
    const int wc = idx & 255;
    int t = idx >> 8;
    const int hh = t & 255; t >>= 8;
    const int rc = t % 12, b = t / 12;
    float acc = b1[rc];
    for (int c = 0; c < 3; ++c)
        #pragma unroll
        for (int dy = 0; dy < 5; ++dy)
            #pragma unroll
            for (int dx = 0; dx < 5; ++dx) {
                const int ys = (hh + dy - 2 + 256) & 255;
                const int xs = (wc + dx - 2 + 256) & 255;
                acc += y[(((size_t)(b * 3 + c)) << 16) + (ys << 8) + xs] *
                       w1[((rc * 3 + c) * 5 + dy) * 5 + dx];
            }
    r1[idx] = acc > 0.f ? acc : 0.01f * acc;
}

__global__ __launch_bounds__(256) void res2_kernel(
    const float* __restrict__ y, const float* __restrict__ r1,
    const float* __restrict__ w2, const float* __restrict__ b2,
    float* __restrict__ out)
{
    const int idx = blockIdx.x * blockDim.x + threadIdx.x;
    if (idx >= BATCH * 3 * 256 * 256) return;
    const int wc = idx & 255;
    int t = idx >> 8;
    const int hh = t & 255; t >>= 8;
    const int c = t % 3, b = t / 3;
    float acc = b2[c];
    #pragma unroll
    for (int rc = 0; rc < 12; ++rc)
        acc += r1[(((size_t)(b * 12 + rc)) << 16) + (hh << 8) + wc] * w2[c * 12 + rc];
    out[idx] = y[idx] + acc;
}

// =====================================================================
extern "C" void kernel_launch(void* const* d_in, const int* in_sizes, int n_in,
                              void* d_out, int out_size, void* d_ws, size_t ws_size,
                              hipStream_t stream)
{
    const float* x         = (const float*)d_in[0];
    const float* pos_w     = (const float*)d_in[1];
    const float* pos_b     = (const float*)d_in[2];
    const float* patch_w   = (const float*)d_in[3];
    const float* patch_b   = (const float*)d_in[4];
    const float* ln1_g     = (const float*)d_in[5];
    const float* ln1_b     = (const float*)d_in[6];
    const float* qkv_w     = (const float*)d_in[7];
    const float* qkv_b     = (const float*)d_in[8];
    const float* rpb       = (const float*)d_in[9];
    const float* proj_w    = (const float*)d_in[10];
    const float* proj_b    = (const float*)d_in[11];
    const float* ln2_g     = (const float*)d_in[12];
    const float* ln2_b     = (const float*)d_in[13];
    const float* fc1_w     = (const float*)d_in[14];
    const float* fc1_b     = (const float*)d_in[15];
    const float* fc2_w     = (const float*)d_in[16];
    const float* fc2_b     = (const float*)d_in[17];
    const float* depatch_w = (const float*)d_in[18];
    const float* depatch_b = (const float*)d_in[19];
    const float* r1w       = (const float*)d_in[20];
    const float* r1b       = (const float*)d_in[21];
    const float* r2w       = (const float*)d_in[22];
    const float* r2b       = (const float*)d_in[23];

    // ---- workspace layout: f32 residual/image buffers + f16 GEMM operands
    char* wsb = (char*)d_ws;
    float*    f    = (float*)wsb;                 wsb += (size_t)NTPAD * 128 * 4;  // residual stream
    float*    ybuf = (float*)wsb;                 wsb += (size_t)BATCH * 3 * 256 * 256 * 4;
    float*    rbuf = (float*)wsb;                 wsb += (size_t)BATCH * 12 * 256 * 256 * 4;
    _Float16* t0   = (_Float16*)wsb;              wsb += (size_t)NAPAD * 128 * 2;  // LN out (padded or not)
    _Float16* big  = (_Float16*)wsb;              wsb += (size_t)NTPAD * 512 * 2;  // qkv (NAPAD*384) / fc1 hidden
    _Float16* aout = (_Float16*)wsb;              wsb += (size_t)NTPAD * 128 * 2;  // attention out
    _Float16* wq   = (_Float16*)wsb;              wsb += (size_t)3 * 384 * 128 * 2; // f16 weights
    _Float16* wp   = (_Float16*)wsb;              wsb += (size_t)3 * 128 * 128 * 2;
    _Float16* w1   = (_Float16*)wsb;              wsb += (size_t)3 * 512 * 128 * 2;
    _Float16* w2   = (_Float16*)wsb;              wsb += (size_t)3 * 128 * 512 * 2;

    // ---- one-time f32 -> f16 weight conversion
    cvt_f16_kernel<<<(3 * 384 * 128 + 255) / 256, 256, 0, stream>>>(qkv_w, wq, 3 * 384 * 128);
    cvt_f16_kernel<<<(3 * 128 * 128 + 255) / 256, 256, 0, stream>>>(proj_w, wp, 3 * 128 * 128);
    cvt_f16_kernel<<<(3 * 512 * 128 + 255) / 256, 256, 0, stream>>>(fc1_w, w1, 3 * 512 * 128);
    cvt_f16_kernel<<<(3 * 128 * 512 + 255) / 256, 256, 0, stream>>>(fc2_w, w2, 3 * 128 * 512);

    // ---- patch embedding
    patch_conv_kernel<<<NTPAD, 128, 0, stream>>>(x, pos_w, pos_b, patch_w, patch_b, f);

    // ---- 3 NAT transformer blocks
    for (int l = 0; l < 3; ++l) {
        ln_kernel<<<NAPAD, 128, 0, stream>>>(f, t0, ln1_g + l * 128, ln1_b + l * 128, 1);
        gemm_wmma_kernel<<<dim3(NAPAD / 32, 3), 64, 0, stream>>>(
            t0, wq + (size_t)l * 384 * 128, qkv_b + l * 384,
            nullptr, big, 128, 384, 0, 0);
        attn_kernel<<<(NT * NHEADS + 255) / 256, 256, 0, stream>>>(big, rpb + l * 648, aout);
        gemm_wmma_kernel<<<dim3(NTPAD / 32, 1), 64, 0, stream>>>(
            aout, wp + (size_t)l * 128 * 128, proj_b + l * 128,
            f, nullptr, 128, 128, 0, 1);
        ln_kernel<<<NTPAD, 128, 0, stream>>>(f, t0, ln2_g + l * 128, ln2_b + l * 128, 0);
        gemm_wmma_kernel<<<dim3(NTPAD / 32, 4), 64, 0, stream>>>(
            t0, w1 + (size_t)l * 512 * 128, fc1_b + l * 512,
            nullptr, big, 128, 512, 1, 0);
        gemm_wmma_kernel<<<dim3(NTPAD / 32, 1), 64, 0, stream>>>(
            big, w2 + (size_t)l * 128 * 512, fc2_b + l * 128,
            f, nullptr, 512, 128, 0, 1);
    }

    // ---- depatchify + residual conv head
    depatch_kernel<<<(BATCH * 3 * 65536 + 255) / 256, 256, 0, stream>>>(f, depatch_w, depatch_b, ybuf);
    res1_kernel<<<(BATCH * 12 * 65536 + 255) / 256, 256, 0, stream>>>(ybuf, r1w, r1b, rbuf);
    res2_kernel<<<(BATCH * 3 * 65536 + 255) / 256, 256, 0, stream>>>(ybuf, rbuf, r2w, r2b, (float*)d_out);
}
